// EncoderLayer_73821897884034
// MI455X (gfx1250) — matrix operand order repaired and checked by reading, not compiled, not run on hardware
//
#include <hip/hip_runtime.h>
#include <hip/hip_bf16.h>

typedef __attribute__((ext_vector_type(16))) __bf16 v16bf;
typedef __attribute__((ext_vector_type(8)))  float  v8f;

// ---------------------------------------------------------------- utilities

static __device__ __forceinline__ v8f wmma_bf16(v16bf a, v16bf b, v8f c) {
  // D = A(16x32 bf16) * B(32x16 bf16) + C(16x16 f32)
  return __builtin_amdgcn_wmma_f32_16x16x32_bf16(false, a, false, b, (short)0, c,
                                                 false, false);
}

static __device__ __forceinline__ __bf16 f2bf(float f) { return (__bf16)f; }

// order-preserving float -> uint key (for atomic max on floats incl. negatives)
static __device__ __forceinline__ unsigned fkey(float f) {
  unsigned u = __float_as_uint(f);
  return (u >> 31) ? ~u : (u | 0x80000000u);
}
static __device__ __forceinline__ float kinv(unsigned k) {
  unsigned u = (k >> 31) ? (k & 0x7FFFFFFFu) : ~k;
  return __uint_as_float(u);
}

// Packed fragment fetch: lane's 16 bf16 contiguous (32 B -> 2x b128 loads).
static __device__ __forceinline__ v16bf frag_packed(const __bf16* wp, int fidx, int lane) {
  return ((const v16bf*)wp)[fidx * 32 + lane];
}

// Cooperative split of a 16x(32*KT) f32 LDS tile into A-fragment-packed bf16
// hi/lo LDS arrays. Done once per block (256 threads), so the fp32->bf16 split
// cost is paid once instead of per-wave. A layout per ISA 7.12.2 (16-bit A):
// lane: m=l&15, g=l>>4 ; elem e: j=e>>1, kloc=(j>>2)*16 + g*8 + (j&3)*2 + (e&1).
template <int KT>
static __device__ __forceinline__ void pack_a_f32(const float (*xn)[128],
                                                  __bf16* __restrict__ aH,
                                                  __bf16* __restrict__ aL, int t) {
  for (int i = t; i < KT * 512; i += 256) {
    const int e = i & 15, l = (i >> 4) & 31, kt = i >> 9;
    const int m = l & 15, g = l >> 4, j = e >> 1;
    const int kloc = ((j >> 2) << 4) + (g << 3) + ((j & 3) << 1) + (e & 1);
    float f = xn[m][kt * 32 + kloc];
    __bf16 h = f2bf(f);
    aH[i] = h;
    aL[i] = f2bf(f - (float)h);
  }
  __syncthreads();
}

// In-place layernorm of a 16x128 f32 LDS tile (Bessel-corrected std, matches ref).
static __device__ __forceinline__ void layernorm16x128(float (*xn)[128], float (*red1)[16],
                                                        float (*red2)[16], float (*stats)[2],
                                                        const float* g, const float* b) {
  const int t = threadIdx.x, r = t >> 4, cl = t & 15;
  float s1 = 0.f, s2 = 0.f;
  #pragma unroll
  for (int k = 0; k < 8; ++k) {
    float v = xn[r][cl + 16 * k];
    s1 += v; s2 += v * v;
  }
  red1[r][cl] = s1; red2[r][cl] = s2;
  __syncthreads();
  if (cl == 0) {
    float a = 0.f, q = 0.f;
    #pragma unroll
    for (int k = 0; k < 16; ++k) { a += red1[r][k]; q += red2[r][k]; }
    float mean = a * (1.f / 128.f);
    float var  = (q - 128.f * mean * mean) * (1.f / 127.f);
    var = var < 0.f ? 0.f : var;
    stats[r][0] = mean;
    stats[r][1] = 1.f / (sqrtf(var) + 1e-6f);
  }
  __syncthreads();
  #pragma unroll
  for (int i = t; i < 2048; i += 256) {
    int rr = i >> 7, cc = i & 127;
    xn[rr][cc] = g[cc] * ((xn[rr][cc] - stats[rr][0]) * stats[rr][1]) + b[cc];
  }
  __syncthreads();
}

// ---------------------------------------------------------------- kernels

// fp32 weight [K][NN] -> bf16 hi/lo in B-fragment-packed layout:
//   fragment f = kt*(NN/16) + ct ; element (lane,e) -> k = kt*32 + (e>>3)*16 + (lane>>4)*8 + (e&7),
//   n = ct*16 + (lane&15). Each lane's 16 bf16 are contiguous (32 B).
__global__ __launch_bounds__(256) void k_pack_bf16(const float* __restrict__ src,
                                                   __bf16* __restrict__ hi,
                                                   __bf16* __restrict__ lo, int K, int NN) {
  const int total = K * NN;
  const int CT = NN >> 4;
  for (int idx = blockIdx.x * blockDim.x + threadIdx.x; idx < total;
       idx += gridDim.x * blockDim.x) {
    const int e = idx & 15;
    const int l = (idx >> 4) & 31;
    const int f = idx >> 9;
    const int kt = f / CT, ct = f - kt * CT;
    const int k = kt * 32 + ((e >> 3) << 4) + ((l >> 4) << 3) + (e & 7);
    const int n = (ct << 4) + (l & 15);
    float v = src[k * NN + n];
    __bf16 h = f2bf(v);
    hi[idx] = h;
    lo[idx] = f2bf(v - (float)h);
  }
}

// fused LN1 + projection GEMM: x = LN1(nf) @ W_att   (16 rows / block, 8 waves = 8 col tiles)
__global__ __launch_bounds__(256) void k_ln1_proj(const float* __restrict__ nf,
                                                  const float* __restrict__ g,
                                                  const float* __restrict__ b,
                                                  const __bf16* __restrict__ wh,
                                                  const __bf16* __restrict__ wl,
                                                  float* __restrict__ xout, int N) {
  __shared__ float  xn[16][128];
  __shared__ __bf16 xaH[4 * 32 * 16];  // A-fragment packed hi
  __shared__ __bf16 xaL[4 * 32 * 16];  // A-fragment packed lo
  __shared__ float  red1[16][16], red2[16][16], stats[16][2];
  const int rowbase = blockIdx.x * 16;
  const int t = threadIdx.x;
  for (int i = t; i < 2048; i += 256) {
    int rr = i >> 7, cc = i & 127;
    int row = rowbase + rr; if (row >= N) row = N - 1;
    xn[rr][cc] = nf[(long long)row * 128 + cc];
  }
  __syncthreads();
  layernorm16x128(xn, red1, red2, stats, g, b);
  pack_a_f32<4>(xn, xaH, xaL, t);

  const int wv = t >> 5, lane = t & 31;  // ct = wv, CT = 8
  v8f c = {};
  #pragma unroll
  for (int s = 0; s < 4; ++s) {
    v16bf ahi = ((const v16bf*)xaH)[s * 32 + lane];
    v16bf alo = ((const v16bf*)xaL)[s * 32 + lane];
    v16bf bhi = frag_packed(wh, s * 8 + wv, lane);
    v16bf blo = frag_packed(wl, s * 8 + wv, lane);
    c = wmma_bf16(ahi, bhi, c);
    c = wmma_bf16(ahi, blo, c);
    c = wmma_bf16(alo, bhi, c);
  }
  const int n = (wv << 4) + (lane & 15);
  #pragma unroll
  for (int r = 0; r < 8; ++r) {
    int row = rowbase + r + ((lane >> 4) << 3);
    if (row < N) xout[(long long)row * 128 + n] = c[r];
  }
}

// per-(node,head) attention score dots:  a_s[n,h] = <x[n,h,:], att_src[h,:]>
__global__ __launch_bounds__(256) void k_attn_scores(const float* __restrict__ x,
                                                     const float* __restrict__ asrc,
                                                     const float* __restrict__ adst,
                                                     float* __restrict__ aS,
                                                     float* __restrict__ aD, int N) {
  int idx = blockIdx.x * blockDim.x + threadIdx.x;
  if (idx >= N * 8) return;
  int n = idx >> 3, h = idx & 7;
  float s1 = 0.f, s2 = 0.f;
  #pragma unroll
  for (int i = 0; i < 16; ++i) {
    float v = x[(long long)n * 128 + h * 16 + i];
    s1 += v * asrc[h * 16 + i];
    s2 += v * adst[h * 16 + i];
  }
  aS[idx] = s1;
  aD[idx] = s2;
}

// edge pass A: segment max (self-loops appended as edges E..E+N-1)
__global__ __launch_bounds__(256) void k_edge_max(const long long* __restrict__ ei,
                                                  const float* __restrict__ aS,
                                                  const float* __restrict__ aD,
                                                  unsigned* __restrict__ mkey, int E, int N) {
  long long total = (long long)(E + N) * 8;
  long long stride = (long long)gridDim.x * blockDim.x;
  for (long long idx = blockIdx.x * (long long)blockDim.x + threadIdx.x; idx < total;
       idx += stride) {
    int e = (int)(idx >> 3), h = (int)(idx & 7);
    int s, d;
    if (e < E) { s = (int)ei[e]; d = (int)ei[E + e]; } else { s = d = e - E; }
    float v = aS[s * 8 + h] + aD[d * 8 + h];
    v = v >= 0.f ? v : 0.2f * v;
    atomicMax(&mkey[d * 8 + h], fkey(v));
  }
}

// edge pass B: softmax denominators
__global__ __launch_bounds__(256) void k_edge_sum(const long long* __restrict__ ei,
                                                  const float* __restrict__ aS,
                                                  const float* __restrict__ aD,
                                                  const unsigned* __restrict__ mkey,
                                                  float* __restrict__ denom, int E, int N) {
  long long total = (long long)(E + N) * 8;
  long long stride = (long long)gridDim.x * blockDim.x;
  for (long long idx = blockIdx.x * (long long)blockDim.x + threadIdx.x; idx < total;
       idx += stride) {
    int e = (int)(idx >> 3), h = (int)(idx & 7);
    int s, d;
    if (e < E) { s = (int)ei[e]; d = (int)ei[E + e]; } else { s = d = e - E; }
    float v = aS[s * 8 + h] + aD[d * 8 + h];
    v = v >= 0.f ? v : 0.2f * v;
    atomicAdd(&denom[d * 8 + h], __expf(v - kinv(mkey[d * 8 + h])));
  }
}

// edge pass C: message scatter (wave per edge; lane covers 4 of 128 cols)
__global__ __launch_bounds__(256) void k_scatter(const long long* __restrict__ ei,
                                                 const float* __restrict__ aS,
                                                 const float* __restrict__ aD,
                                                 const unsigned* __restrict__ mkey,
                                                 const float* __restrict__ denom,
                                                 const float* __restrict__ x,
                                                 float* __restrict__ gat, int E, int N) {
  const int lane = threadIdx.x & 31;
  const int wid = (blockIdx.x * blockDim.x + threadIdx.x) >> 5;
  const int nw = (gridDim.x * blockDim.x) >> 5;
  const int total = E + N;
  for (int e = wid; e < total; e += nw) {
    int s, d;
    if (e < E) { s = (int)ei[e]; d = (int)ei[E + e]; } else { s = d = e - E; }
    #pragma unroll
    for (int k = 0; k < 4; ++k) {
      int col = lane + 32 * k;
      int h = col >> 4;
      float v = aS[s * 8 + h] + aD[d * 8 + h];
      v = v >= 0.f ? v : 0.2f * v;
      float coef = __expf(v - kinv(mkey[d * 8 + h])) / (denom[d * 8 + h] + 1e-16f);
      atomicAdd(&gat[(long long)d * 128 + col], x[(long long)s * 128 + col] * coef);
    }
  }
}

// residual: out = nf + gat + bias_att   (out becomes the FFN input "nf2")
__global__ __launch_bounds__(256) void k_add(const float* __restrict__ nf,
                                             const float* __restrict__ gat,
                                             const float* __restrict__ bias,
                                             float* __restrict__ out, long long total) {
  long long stride = (long long)gridDim.x * blockDim.x;
  for (long long i = blockIdx.x * (long long)blockDim.x + threadIdx.x; i < total; i += stride)
    out[i] = nf[i] + gat[i] + bias[i & 127];
}

// fully fused LN2 + FFN:  io = io + relu(LN2(io) @ W1 + b1) @ W2 + b2
// Hidden (16x512) lives in LDS in A-fragment-packed bf16 hi/lo -- never touches HBM.
__global__ __launch_bounds__(256) void k_ffn(float* __restrict__ io,
                                             const float* __restrict__ g,
                                             const float* __restrict__ b,
                                             const __bf16* __restrict__ w1h,
                                             const __bf16* __restrict__ w1l,
                                             const float* __restrict__ b1,
                                             const __bf16* __restrict__ w2h,
                                             const __bf16* __restrict__ w2l,
                                             const float* __restrict__ b2, int N) {
  __shared__ float  res[16][128];
  __shared__ float  xn[16][128];
  __shared__ __bf16 xaH[4 * 32 * 16];   // LN2 output, A-fragment packed hi
  __shared__ __bf16 xaL[4 * 32 * 16];   //                               lo
  __shared__ __bf16 hhA[16 * 32 * 16];  // hidden, A-fragment packed hi
  __shared__ __bf16 hlA[16 * 32 * 16];  //                            lo
  __shared__ float  red1[16][16], red2[16][16], stats[16][2];

  const int rowbase = blockIdx.x * 16;
  const int t = threadIdx.x;
  for (int i = t; i < 2048; i += 256) {
    int rr = i >> 7, cc = i & 127;
    int row = rowbase + rr; if (row >= N) row = N - 1;
    float v = io[(long long)row * 128 + cc];
    res[rr][cc] = v;
    xn[rr][cc] = v;
  }
  __syncthreads();
  layernorm16x128(xn, red1, red2, stats, g, b);
  pack_a_f32<4>(xn, xaH, xaL, t);

  const int wv = t >> 5, lane = t & 31;

  // A fragments live in registers across the whole GEMM1 col loop.
  v16bf aH[4], aL[4];
  #pragma unroll
  for (int s = 0; s < 4; ++s) {
    aH[s] = ((const v16bf*)xaH)[s * 32 + lane];
    aL[s] = ((const v16bf*)xaL)[s * 32 + lane];
  }

  // GEMM1: [16x128] x [128x512] -> relu -> LDS (A-fragment layout for GEMM2)
  #pragma unroll
  for (int q = 0; q < 4; ++q) {
    const int ct = wv + 8 * q;           // col tile of W1 (CT = 32)
    v8f c = {};
    #pragma unroll
    for (int s = 0; s < 4; ++s) {
      v16bf bhi = frag_packed(w1h, s * 32 + ct, lane);
      v16bf blo = frag_packed(w1l, s * 32 + ct, lane);
      c = wmma_bf16(aH[s], bhi, c);
      c = wmma_bf16(aH[s], blo, c);
      c = wmma_bf16(aL[s], bhi, c);
    }
    const int n = (ct << 4) + (lane & 15);  // hidden col = K index of GEMM2
    const int kt2 = n >> 5;
    const int kin = n & 31;
    // invert A-layout: kloc = (j>>2)*16 + g*8 + (j&3)*2 + (e&1)  => lane/elem
    const int half = kin >> 4;
    const int rem = kin & 15;
    const int gg = rem >> 3;
    const int within = rem & 7;
    const int e = ((half << 2) + (within >> 1)) * 2 + (within & 1);
    #pragma unroll
    for (int r = 0; r < 8; ++r) {
      const int m = r + ((lane >> 4) << 3);
      const int idx = ((kt2 << 5) + (gg << 4) + m) * 16 + e;
      float hval = c[r] + b1[n];
      hval = hval > 0.f ? hval : 0.f;
      __bf16 h = f2bf(hval);
      hhA[idx] = h;
      hlA[idx] = f2bf(hval - (float)h);
    }
  }
  __syncthreads();

  // GEMM2: [16x512] x [512x128] + residual + b2
  v8f c = {};
  #pragma unroll
  for (int s = 0; s < 16; ++s) {
    v16bf ahi = ((const v16bf*)hhA)[s * 32 + lane];
    v16bf alo = ((const v16bf*)hlA)[s * 32 + lane];
    v16bf bhi = frag_packed(w2h, s * 8 + wv, lane);
    v16bf blo = frag_packed(w2l, s * 8 + wv, lane);
    c = wmma_bf16(ahi, bhi, c);
    c = wmma_bf16(ahi, blo, c);
    c = wmma_bf16(alo, bhi, c);
  }
  const int n = (wv << 4) + (lane & 15);
  #pragma unroll
  for (int r = 0; r < 8; ++r) {
    int m = r + ((lane >> 4) << 3);
    int row = rowbase + m;
    if (row < N) io[(long long)row * 128 + n] = res[m][n] + c[r] + b2[n];
  }
}

// ---------------------------------------------------------------- launcher

extern "C" void kernel_launch(void* const* d_in, const int* in_sizes, int n_in,
                              void* d_out, int out_size, void* d_ws, size_t ws_size,
                              hipStream_t stream) {
  const float*     nf       = (const float*)d_in[0];
  const long long* ei       = (const long long*)d_in[1];  // int64 [2,E]
  const float*     ln1_g    = (const float*)d_in[2];
  const float*     ln1_b    = (const float*)d_in[3];
  const float*     W_att    = (const float*)d_in[4];
  const float*     att_src  = (const float*)d_in[5];
  const float*     att_dst  = (const float*)d_in[6];
  const float*     bias_att = (const float*)d_in[7];
  const float*     ln2_g    = (const float*)d_in[8];
  const float*     ln2_b    = (const float*)d_in[9];
  const float*     W1       = (const float*)d_in[10];
  const float*     b1       = (const float*)d_in[11];
  const float*     W2       = (const float*)d_in[12];
  const float*     b2       = (const float*)d_in[13];

  const int N = in_sizes[0] / 128;
  const int E = in_sizes[1] / 2;
  float* out = (float*)d_out;

  // workspace carve (256B aligned)
  char* p = (char*)d_ws;
  auto take = [&p](size_t bytes) {
    char* r = p;
    p += (bytes + 255) & ~(size_t)255;
    return r;
  };
  float*    x     = (float*)take((size_t)N * 128 * 4);
  float*    gat   = (float*)take((size_t)N * 128 * 4);
  float*    aS    = (float*)take((size_t)N * 8 * 4);
  float*    aD    = (float*)take((size_t)N * 8 * 4);
  unsigned* mkey  = (unsigned*)take((size_t)N * 8 * 4);
  float*    denom = (float*)take((size_t)N * 8 * 4);
  __bf16*   wAh   = (__bf16*)take((size_t)128 * 128 * 2);
  __bf16*   wAl   = (__bf16*)take((size_t)128 * 128 * 2);
  __bf16*   w1h   = (__bf16*)take((size_t)128 * 512 * 2);
  __bf16*   w1l   = (__bf16*)take((size_t)128 * 512 * 2);
  __bf16*   w2h   = (__bf16*)take((size_t)512 * 128 * 2);
  __bf16*   w2l   = (__bf16*)take((size_t)512 * 128 * 2);

  hipMemsetAsync(gat, 0, (size_t)N * 128 * 4, stream);
  hipMemsetAsync(mkey, 0, (size_t)N * 8 * 4, stream);   // key 0 < any real key
  hipMemsetAsync(denom, 0, (size_t)N * 8 * 4, stream);

  k_pack_bf16<<<64, 256, 0, stream>>>(W_att, wAh, wAl, 128, 128);
  k_pack_bf16<<<256, 256, 0, stream>>>(W1, w1h, w1l, 128, 512);
  k_pack_bf16<<<256, 256, 0, stream>>>(W2, w2h, w2l, 512, 128);

  const int nblk = (N + 15) / 16;
  k_ln1_proj<<<nblk, 256, 0, stream>>>(nf, ln1_g, ln1_b, wAh, wAl, x, N);
  k_attn_scores<<<(N * 8 + 255) / 256, 256, 0, stream>>>(x, att_src, att_dst, aS, aD, N);
  k_edge_max<<<2048, 256, 0, stream>>>(ei, aS, aD, mkey, E, N);
  k_edge_sum<<<2048, 256, 0, stream>>>(ei, aS, aD, mkey, denom, E, N);
  k_scatter<<<8192, 256, 0, stream>>>(ei, aS, aD, mkey, denom, x, gat, E, N);

  const long long tot = (long long)N * 128;
  k_add<<<(int)((tot + 255) / 256), 256, 0, stream>>>(nf, gat, bias_att, out, tot);
  k_ffn<<<nblk, 256, 0, stream>>>(out, ln2_g, ln2_b, w1h, w1l, b1, w2h, w2l, b2, N);
}
